// SimpleDeepseekV3MoEModel_11802570130394
// MI455X (gfx1250) — matrix operand order repaired and checked by reading, not compiled, not run on hardware
//
#include <hip/hip_runtime.h>
#include <math.h>

// ---------------- model constants ----------------
constexpr int S  = 2048;   // sequence
constexpr int Dm = 768;    // model dim
constexpr int Hh = 12;     // heads (Dh = 64)
constexpr int Ff = 1536;   // expert hidden
constexpr int Ee = 8;      // experts
constexpr int Ll = 2;      // layers
constexpr int Vv = 32000;  // vocab

typedef __attribute__((ext_vector_type(16))) __bf16 v16bf;
typedef __attribute__((ext_vector_type(8)))  float  v8f;
typedef unsigned int u32x4 __attribute__((ext_vector_type(4)));
typedef int          i32x4 __attribute__((ext_vector_type(4)));
typedef int          i32x8 __attribute__((ext_vector_type(8)));

__device__ inline __bf16 f2bf(float x) { return (__bf16)x; }

// ---------------- WMMA fragment loaders (layouts per CDNA5 ISA) ----------------
// A matrix 16x32 bf16: lanes 0-15 carry rows 0-15 with K=0..7,16..23; lanes 16-31 same rows, K=8..15,24..31.
__device__ inline v16bf load_a_frag(const __bf16* A, int lda, int row0, int k0, int lane) {
  int m  = lane & 15;
  int kb = (lane < 16) ? 0 : 8;
  const __bf16* p = A + (size_t)(row0 + m) * lda + k0;
  v16bf a;
#pragma unroll
  for (int i = 0; i < 8; ++i) {
    int k = ((i < 4) ? 0 : 16) + kb + (i & 3) * 2;
    a[2 * i]     = p[k];
    a[2 * i + 1] = p[k + 1];
  }
  return a;
}

__device__ inline v16bf load_a_frag_f32(const float* A, int lda, int row0, int k0, int lane) {
  int m  = lane & 15;
  int kb = (lane < 16) ? 0 : 8;
  const float* p = A + (size_t)(row0 + m) * lda + k0;
  v16bf a;
#pragma unroll
  for (int i = 0; i < 8; ++i) {
    int k = ((i < 4) ? 0 : 16) + kb + (i & 3) * 2;
    a[2 * i]     = f2bf(p[k]);
    a[2 * i + 1] = f2bf(p[k + 1]);
  }
  return a;
}

// A fragment with per-row gather (MoE token grouping): row m comes from tok[m].
__device__ inline v16bf load_a_frag_gather(const __bf16* A, int lda, const int* tok, int k0, int lane) {
  int m  = lane & 15;
  int kb = (lane < 16) ? 0 : 8;
  const __bf16* p = A + (size_t)tok[m] * lda + k0;
  v16bf a;
#pragma unroll
  for (int i = 0; i < 8; ++i) {
    int k = ((i < 4) ? 0 : 16) + kb + (i & 3) * 2;
    a[2 * i]     = p[k];
    a[2 * i + 1] = p[k + 1];
  }
  return a;
}

// B matrix 32x16 (KxN) built from row-major W[N][K] (i.e. B = W^T):
// lanes 0-15 carry K=0..7,16..23, lanes 16-31 carry K=8..15,24..31; column = lane&15.
__device__ inline v16bf load_b_frag_nk(const __bf16* W, int ldb, int n0, int k0, int lane) {
  int n  = n0 + (lane & 15);
  int kb = k0 + ((lane < 16) ? 0 : 8);
  const __bf16* p = W + (size_t)n * ldb + kb;
  v16bf b;
#pragma unroll
  for (int e = 0; e < 16; ++e) b[e] = p[(e < 8) ? e : e + 8];
  return b;
}

// B from memory already stored KxN row-major (e.g. V matrix: keys x headdim).
__device__ inline v16bf load_b_frag_kn(const __bf16* M, int ldm, int k0, int n0, int lane) {
  int n  = n0 + (lane & 15);
  int kb = k0 + ((lane < 16) ? 0 : 8);
  v16bf b;
#pragma unroll
  for (int e = 0; e < 16; ++e) b[e] = M[(size_t)(kb + ((e < 8) ? e : e + 8)) * ldm + n];
  return b;
}

__device__ inline v8f wmma_bf16(v16bf a, v16bf b, v8f c) {
  // (neg_a, A, neg_b, B, c_mod, C, reuse_a, reuse_b)
  return __builtin_amdgcn_wmma_f32_16x16x32_bf16(false, a, false, b, (short)0, c, false, false);
}

// ---------------- Tensor Data Mover: stage a 2-D bf16 tile (rows x rowlen) into LDS ----------------
// D# layout per CDNA5 ISA. 2-D tensor -> groups 2/3 zero.
__device__ inline void tdm_load_tile_bf16(unsigned lds_off, const __bf16* gsrc,
                                          int rowlen, int rows, long long row_stride_elems) {
  unsigned long long ga = (unsigned long long)(size_t)gsrc;
  u32x4 g0;
  g0.x = 1u;                                   // count=1 (valid user descriptor)
  g0.y = lds_off;                              // lds_addr (bytes)
  g0.z = (unsigned)(ga & 0xffffffffull);       // global_addr[31:0]
  g0.w = (unsigned)((ga >> 32) & 0x1ffffffull) // global_addr[56:32]
         | (2u << 30);                         // type=2 ("image")
  i32x8 g1;
  g1[0] = (int)(1u << 16);                                       // data_size=1 -> 2 bytes
  g1[1] = (int)(((unsigned)rowlen & 0xffffu) << 16);             // tensor_dim0[15:0]
  g1[2] = (int)((((unsigned)rowlen >> 16) & 0xffffu)             // tensor_dim0[31:16]
                | (((unsigned)rows & 0xffffu) << 16));           // tensor_dim1[15:0]
  g1[3] = (int)(((unsigned)rows >> 16)                           // tensor_dim1[31:16]
                | (((unsigned)rowlen & 0xffffu) << 16));         // tile_dim0
  g1[4] = (int)((unsigned)rows & 0xffffu);                       // tile_dim1 (tile_dim2=0)
  g1[5] = (int)((unsigned long long)row_stride_elems & 0xffffffffull);  // dim0_stride[31:0]
  g1[6] = (int)(((unsigned long long)row_stride_elems >> 32) & 0xffffull); // dim0_stride[47:32]
  g1[7] = 0;
  i32x4 z4 = {0, 0, 0, 0};
#if defined(__clang_major__) && (__clang_major__ >= 23)
  i32x8 z8 = {0, 0, 0, 0, 0, 0, 0, 0};
  __builtin_amdgcn_tensor_load_to_lds(g0, g1, z4, z4, z8, 0);
#else
  __builtin_amdgcn_tensor_load_to_lds(g0, g1, z4, z4, 0);
#endif
}

// ---------------- small helpers ----------------
__device__ inline float block_reduce_sum(float v) {
  __shared__ float red[8];
#pragma unroll
  for (int off = 16; off > 0; off >>= 1) v += __shfl_xor(v, off);
  int lane = threadIdx.x & 31, wv = threadIdx.x >> 5;
  if (lane == 0) red[wv] = v;
  __syncthreads();
  float t = (threadIdx.x < 8) ? red[threadIdx.x] : 0.f;
  if (wv == 0) {
#pragma unroll
    for (int off = 4; off > 0; off >>= 1) t += __shfl_xor(t, off);
    if (lane == 0) red[0] = t;
  }
  __syncthreads();
  float out = red[0];
  __syncthreads();
  return out;
}

// ---------------- kernels ----------------
__global__ void f2bf_kernel(const float* __restrict__ src, __bf16* __restrict__ dst, size_t n) {
  size_t i = (size_t)blockIdx.x * blockDim.x + threadIdx.x;
  size_t st = (size_t)gridDim.x * blockDim.x;
  for (; i < n; i += st) dst[i] = f2bf(src[i]);
}

__global__ void embed_kernel(const int* __restrict__ tokens, const float* __restrict__ emb,
                             float* __restrict__ x, __bf16* __restrict__ xb) {
  int row = blockIdx.x;
  const float* er = emb + (size_t)tokens[row] * Dm;
  const float sc = 27.712812921102035f;  // sqrt(768)
  for (int d = threadIdx.x; d < Dm; d += 256) {
    float v = er[d] * sc;
    x[(size_t)row * Dm + d]  = v;
    xb[(size_t)row * Dm + d] = f2bf(v);
  }
}

// C[MxN] = A(bf16/f32, MxK) * W(bf16, NxK)^T + bias; 64x64 block tile, 8 waves (4M x 2N), wave = 16x32.
// OUT_BF16: fold bias and emit bf16 (used for QKV so attention consumes bf16 directly).
template <bool A_F32, bool OUT_BF16>
__global__ __launch_bounds__(256) void gemm_kernel(const void* __restrict__ Aptr,
                                                   const __bf16* __restrict__ W,
                                                   const float* __restrict__ bias,
                                                   void* __restrict__ Cv, int M, int N, int Kd) {
  int lane = threadIdx.x & 31, wave = threadIdx.x >> 5;
  int wm = wave & 3, wn = wave >> 2;
  int row0 = (blockIdx.y * 4 + wm) * 16;
  int col0 = (blockIdx.x * 2 + wn) * 32;
  if (row0 >= M || col0 >= N) return;
  v8f acc0 = {}, acc1 = {};
  for (int k = 0; k < Kd; k += 32) {
    v16bf a;
    if constexpr (A_F32) a = load_a_frag_f32((const float*)Aptr, Kd, row0, k, lane);
    else                 a = load_a_frag((const __bf16*)Aptr, Kd, row0, k, lane);
    if (k + 32 < Kd)
      __builtin_prefetch(W + (size_t)(col0 + (lane & 15)) * Kd + k + 32, 0, 0);
    acc0 = wmma_bf16(a, load_b_frag_nk(W, Kd, col0, k, lane), acc0);
    acc1 = wmma_bf16(a, load_b_frag_nk(W, Kd, col0 + 16, k, lane), acc1);
  }
#pragma unroll
  for (int r = 0; r < 8; ++r) {
    int row = row0 + r + ((lane < 16) ? 0 : 8);
    int c0  = col0 + (lane & 15);
    float b0 = bias ? bias[c0] : 0.f;
    float b1 = bias ? bias[c0 + 16] : 0.f;
    float o0 = acc0[r] + b0, o1 = acc1[r] + b1;
    if constexpr (OUT_BF16) {
      ((__bf16*)Cv)[(size_t)row * N + c0]      = f2bf(o0);
      ((__bf16*)Cv)[(size_t)row * N + c0 + 16] = f2bf(o1);
    } else {
      ((float*)Cv)[(size_t)row * N + c0]      = o0;
      ((float*)Cv)[(size_t)row * N + c0 + 16] = o1;
    }
  }
}

// LM head: C[S x Vv] = xb @ W^T, W tile (64 rows x 768) staged into LDS by the Tensor Data Mover.
__global__ __launch_bounds__(256) void lmhead_kernel(const __bf16* __restrict__ xb,
                                                     const __bf16* __restrict__ W,
                                                     float* __restrict__ C) {
  extern __shared__ __bf16 wtile[];  // 64 * 768 bf16 = 96 KB dynamic LDS
  int lane = threadIdx.x & 31, wave = threadIdx.x >> 5;
  int n0   = blockIdx.x * 64;
  int row0 = blockIdx.y * 64 + (wave & 3) * 16;
  int coff = (wave >> 2) * 32;

  if (wave == 0) {
    unsigned lds_off = (unsigned)(size_t)(void*)wtile;  // generic LDS ptr low 32 bits = LDS offset
    tdm_load_tile_bf16(lds_off, W + (size_t)n0 * Dm, Dm, 64, Dm);
    __builtin_amdgcn_s_wait_tensorcnt(0);
  }
  __syncthreads();

  v8f acc0 = {}, acc1 = {};
  for (int k = 0; k < Dm; k += 32) {
    v16bf a = load_a_frag(xb, Dm, row0, k, lane);
    acc0 = wmma_bf16(a, load_b_frag_nk(wtile, Dm, coff, k, lane), acc0);
    acc1 = wmma_bf16(a, load_b_frag_nk(wtile, Dm, coff + 16, k, lane), acc1);
  }
#pragma unroll
  for (int r = 0; r < 8; ++r) {
    int row = row0 + r + ((lane < 16) ? 0 : 8);
    int c   = n0 + coff + (lane & 15);
    C[(size_t)row * Vv + c]      = acc0[r];
    C[(size_t)row * Vv + c + 16] = acc1[r];
  }
}

// Flash-style attention over bf16 QKV: one wave per (16-query tile, head). Dh=64 fixed.
__global__ __launch_bounds__(256) void attn_kernel(const __bf16* __restrict__ qkvb,
                                                   float* __restrict__ obuf) {
  __shared__ __bf16 pbuf[8][16 * 32];
  int lane = threadIdx.x & 31, wave = threadIdx.x >> 5;
  int wid = blockIdx.x * 8 + wave;
  int qt = wid % (S / 16);
  int h  = wid / (S / 16);
  const __bf16* qbase = qkvb + h * 64;
  const __bf16* kbase = qkvb + Dm + h * 64;
  const __bf16* vbase = qkvb + 2 * Dm + h * 64;
  const int ldq = 3 * Dm;

  v16bf aq0 = load_a_frag(qbase, ldq, qt * 16, 0, lane);
  v16bf aq1 = load_a_frag(qbase, ldq, qt * 16, 32, lane);
  v8f o0 = {}, o1 = {}, o2 = {}, o3 = {};
  float m_i[8], l_i[8];
#pragma unroll
  for (int r = 0; r < 8; ++r) { m_i[r] = -3.0e38f; l_i[r] = 0.f; }
  const float sc = 0.125f;  // 1/sqrt(64)

  for (int kt = 0; kt < S; kt += 32) {
    v8f z0 = {}, z1 = {};
    z0 = wmma_bf16(aq0, load_b_frag_nk(kbase, ldq, kt, 0, lane), z0);
    z0 = wmma_bf16(aq1, load_b_frag_nk(kbase, ldq, kt, 32, lane), z0);
    z1 = wmma_bf16(aq0, load_b_frag_nk(kbase, ldq, kt + 16, 0, lane), z1);
    z1 = wmma_bf16(aq1, load_b_frag_nk(kbase, ldq, kt + 16, 32, lane), z1);
#pragma unroll
    for (int r = 0; r < 8; ++r) {
      float sa = z0[r] * sc, sb = z1[r] * sc;
      float cur = fmaxf(sa, sb);
#pragma unroll
      for (int off = 8; off > 0; off >>= 1) cur = fmaxf(cur, __shfl_xor(cur, off));
      float mn = fmaxf(m_i[r], cur);
      float corr = __expf(m_i[r] - mn);
      float p0 = __expf(sa - mn), p1 = __expf(sb - mn);
      float ls = p0 + p1;
#pragma unroll
      for (int off = 8; off > 0; off >>= 1) ls += __shfl_xor(ls, off);
      l_i[r] = l_i[r] * corr + ls;
      m_i[r] = mn;
      o0[r] *= corr; o1[r] *= corr; o2[r] *= corr; o3[r] *= corr;
      int row = r + ((lane < 16) ? 0 : 8);
      pbuf[wave][row * 32 + (lane & 15)]      = f2bf(p0);
      pbuf[wave][row * 32 + 16 + (lane & 15)] = f2bf(p1);
    }
    __builtin_amdgcn_wave_barrier();  // C-layout -> A-layout transpose through wave-private LDS
    v16bf ap = load_a_frag(&pbuf[wave][0], 32, 0, 0, lane);
    o0 = wmma_bf16(ap, load_b_frag_kn(vbase, ldq, kt, 0, lane), o0);
    o1 = wmma_bf16(ap, load_b_frag_kn(vbase, ldq, kt, 16, lane), o1);
    o2 = wmma_bf16(ap, load_b_frag_kn(vbase, ldq, kt, 32, lane), o2);
    o3 = wmma_bf16(ap, load_b_frag_kn(vbase, ldq, kt, 48, lane), o3);
  }
#pragma unroll
  for (int r = 0; r < 8; ++r) {
    float inv = 1.f / l_i[r];
    int row = qt * 16 + r + ((lane < 16) ? 0 : 8);
    float* orow = obuf + (size_t)row * Dm + h * 64 + (lane & 15);
    orow[0]  = o0[r] * inv;
    orow[16] = o1[r] * inv;
    orow[32] = o2[r] * inv;
    orow[48] = o3[r] * inv;
  }
}

// x = LN(x + add); also emits bf16 copy for the next GEMM. One block per row.
__global__ __launch_bounds__(256) void add_ln_kernel(float* __restrict__ x, const float* __restrict__ add,
                                                     const float* __restrict__ w, const float* __restrict__ b,
                                                     __bf16* __restrict__ xb) {
  int row = blockIdx.x;
  float* xr = x + (size_t)row * Dm;
  const float* ar = add + (size_t)row * Dm;
  float v[3];
  float lsum = 0.f;
#pragma unroll
  for (int i = 0; i < 3; ++i) {
    int d = threadIdx.x + i * 256;
    v[i] = xr[d] + ar[d];
    lsum += v[i];
  }
  float mean = block_reduce_sum(lsum) * (1.f / Dm);
  float lvar = 0.f;
#pragma unroll
  for (int i = 0; i < 3; ++i) { float t = v[i] - mean; lvar += t * t; }
  float rstd = rsqrtf(block_reduce_sum(lvar) * (1.f / Dm) + 1e-5f);
#pragma unroll
  for (int i = 0; i < 3; ++i) {
    int d = threadIdx.x + i * 256;
    float o = (v[i] - mean) * rstd * w[d] + b[d];
    xr[d] = o;
    xb[(size_t)row * Dm + d] = f2bf(o);
  }
}

// Final RMSNorm -> bf16 activations for the LM head.
__global__ __launch_bounds__(256) void rms_kernel(const float* __restrict__ x, const float* __restrict__ w,
                                                  __bf16* __restrict__ xb) {
  int row = blockIdx.x;
  const float* xr = x + (size_t)row * Dm;
  float v[3];
  float ss = 0.f;
#pragma unroll
  for (int i = 0; i < 3; ++i) {
    v[i] = xr[threadIdx.x + i * 256];
    ss += v[i] * v[i];
  }
  float r = rsqrtf(block_reduce_sum(ss) * (1.f / Dm) + 1.1920929e-07f);
#pragma unroll
  for (int i = 0; i < 3; ++i) {
    int d = threadIdx.x + i * 256;
    xb[(size_t)row * Dm + d] = f2bf(v[i] * r * w[d]);
  }
}

// Gate: one wave per token; 8 expert dots, top-2, softmax of top-2.
__global__ __launch_bounds__(256) void gate_topk_kernel(const float* __restrict__ x,
                                                        const float* __restrict__ gw,
                                                        int* __restrict__ topi, float* __restrict__ topw) {
  int lane = threadIdx.x & 31, wave = threadIdx.x >> 5;
  int t = blockIdx.x * 8 + wave;
  const float* xr = x + (size_t)t * Dm;
  float s[Ee];
#pragma unroll
  for (int e = 0; e < Ee; ++e) {
    float acc = 0.f;
    for (int d = lane; d < Dm; d += 32) acc += xr[d] * gw[e * Dm + d];
#pragma unroll
    for (int off = 16; off > 0; off >>= 1) acc += __shfl_xor(acc, off);
    s[e] = acc;
  }
  int i0 = 0; float v0 = s[0];
#pragma unroll
  for (int e = 1; e < Ee; ++e) if (s[e] > v0) { v0 = s[e]; i0 = e; }
  int i1 = (i0 == 0) ? 1 : 0; float v1 = s[i1];
#pragma unroll
  for (int e = 0; e < Ee; ++e) if (e != i0 && s[e] > v1) { v1 = s[e]; i1 = e; }
  float e1 = __expf(v1 - v0);
  float inv = 1.f / (1.f + e1);
  if (lane == 0) {
    topi[2 * t] = i0; topi[2 * t + 1] = i1;
    topw[2 * t] = inv; topw[2 * t + 1] = e1 * inv;
  }
}

// Deterministic per-expert token lists via in-block scan (no atomic ordering nondeterminism).
__global__ __launch_bounds__(256) void build_lists_kernel(const int* __restrict__ topi,
                                                          const float* __restrict__ topw,
                                                          int* __restrict__ list, float* __restrict__ listw,
                                                          int* __restrict__ count) {
  int e = blockIdx.x;
  __shared__ int scan[256];
  __shared__ int basewg;
  if (threadIdx.x == 0) basewg = 0;
  __syncthreads();
  for (int c = 0; c < S; c += 256) {
    int t = c + threadIdx.x;
    int flag = 0; float w = 0.f;
    if (topi[2 * t] == e)          { flag = 1; w = topw[2 * t]; }
    else if (topi[2 * t + 1] == e) { flag = 1; w = topw[2 * t + 1]; }
    scan[threadIdx.x] = flag;
    __syncthreads();
    for (int off = 1; off < 256; off <<= 1) {
      int v = (threadIdx.x >= off) ? scan[threadIdx.x - off] : 0;
      __syncthreads();
      scan[threadIdx.x] += v;
      __syncthreads();
    }
    if (flag) {
      int pos = basewg + scan[threadIdx.x] - 1;
      list[e * S + pos] = t;
      listw[e * S + pos] = w;
    }
    __syncthreads();
    if (threadIdx.x == 0) basewg += scan[255];
    __syncthreads();
  }
  if (threadIdx.x == 0) count[e] = basewg;
}

// Grouped-GEMM expert FFN: block = (16-token tile, expert); 8 waves.
// hg = silu(X Wg^T), hu = X Wu^T, h = hg*hu staged in LDS, then h Wd^T scattered with gate weights.
__global__ __launch_bounds__(256) void moe_ffn_kernel(const __bf16* __restrict__ xb,
                                                      const __bf16* __restrict__ egw,
                                                      const __bf16* __restrict__ euw,
                                                      const __bf16* __restrict__ edw,
                                                      const int* __restrict__ list,
                                                      const float* __restrict__ listw,
                                                      const int* __restrict__ count,
                                                      float* __restrict__ moe) {
  int e = blockIdx.y;
  int base = blockIdx.x * 16;
  int cnt = count[e];
  if (base >= cnt) return;
  __shared__ __bf16 hbuf[16 * Ff];  // 48 KB
  __shared__ int tok[16];
  __shared__ float wt[16];
  if (threadIdx.x < 16) {
    int i = base + threadIdx.x;
    tok[threadIdx.x] = (i < cnt) ? list[e * S + i] : 0;
    wt[threadIdx.x]  = (i < cnt) ? listw[e * S + i] : 0.f;
  }
  __syncthreads();
  int lane = threadIdx.x & 31, wave = threadIdx.x >> 5;
  const __bf16* eg = egw + (size_t)e * Ff * Dm;
  const __bf16* eu = euw + (size_t)e * Ff * Dm;
  const __bf16* ed = edw + (size_t)e * Dm * Ff;

  for (int ft = 0; ft < (Ff / 8) / 16; ++ft) {  // 12 tiles of 16 cols per wave
    int col0 = wave * (Ff / 8) + ft * 16;
    v8f cg = {}, cu = {};
    for (int k = 0; k < Dm; k += 32) {
      v16bf a = load_a_frag_gather(xb, Dm, tok, k, lane);
      cg = wmma_bf16(a, load_b_frag_nk(eg, Dm, col0, k, lane), cg);
      cu = wmma_bf16(a, load_b_frag_nk(eu, Dm, col0, k, lane), cu);
    }
#pragma unroll
    for (int r = 0; r < 8; ++r) {
      float g = cg[r];
      float hv = (g / (1.f + __expf(-g))) * cu[r];  // silu(g)*u
      int row = r + ((lane < 16) ? 0 : 8);
      hbuf[row * Ff + col0 + (lane & 15)] = f2bf(hv);
    }
  }
  __syncthreads();
  for (int dt = 0; dt < (Dm / 8) / 16; ++dt) {  // 6 tiles of 16 cols per wave
    int col0 = wave * (Dm / 8) + dt * 16;
    v8f cd = {};
    for (int k = 0; k < Ff; k += 32) {
      v16bf a = load_a_frag(hbuf, Ff, 0, k, lane);
      cd = wmma_bf16(a, load_b_frag_nk(ed, Ff, col0, k, lane), cd);
    }
#pragma unroll
    for (int r = 0; r < 8; ++r) {
      int i = r + ((lane < 16) ? 0 : 8);
      if (base + i < cnt)
        atomicAdd(&moe[(size_t)tok[i] * Dm + col0 + (lane & 15)], cd[r] * wt[i]);
    }
  }
}

// ---------------- host orchestration ----------------
static inline void* wsalloc(char*& p, size_t bytes) {
  void* r = (void*)p;
  p += (bytes + 255) & ~(size_t)255;
  return r;
}

extern "C" void kernel_launch(void* const* d_in, const int* in_sizes, int n_in,
                              void* d_out, int out_size, void* d_ws, size_t ws_size,
                              hipStream_t stream) {
  (void)in_sizes; (void)n_in; (void)out_size; (void)ws_size;
  const int*   tokens = (const int*)d_in[0];
  const float* emb    = (const float*)d_in[1];
  const float* qkv_w  = (const float*)d_in[2];
  const float* qkv_b  = (const float*)d_in[3];
  const float* out_w  = (const float*)d_in[4];
  const float* out_b  = (const float*)d_in[5];
  const float* ln1_w  = (const float*)d_in[6];
  const float* ln1_b  = (const float*)d_in[7];
  const float* ln2_w  = (const float*)d_in[8];
  const float* ln2_b  = (const float*)d_in[9];
  const float* gate_w = (const float*)d_in[10];
  const float* eg_w   = (const float*)d_in[11];
  const float* eu_w   = (const float*)d_in[12];
  const float* ed_w   = (const float*)d_in[13];
  const float* rms_w  = (const float*)d_in[14];
  float* logits = (float*)d_out;

  char* ws = (char*)d_ws;
  const size_t nEmb  = (size_t)Vv * Dm;
  const size_t nQkvW = (size_t)Ll * 3 * Dm * Dm;
  const size_t nOutW = (size_t)Ll * Dm * Dm;
  const size_t nExpW = (size_t)Ll * Ee * Ff * Dm;

  __bf16* emb_bf  = (__bf16*)wsalloc(ws, nEmb * 2);
  __bf16* qkvw_bf = (__bf16*)wsalloc(ws, nQkvW * 2);
  __bf16* outw_bf = (__bf16*)wsalloc(ws, nOutW * 2);
  __bf16* egw_bf  = (__bf16*)wsalloc(ws, nExpW * 2);
  __bf16* euw_bf  = (__bf16*)wsalloc(ws, nExpW * 2);
  __bf16* edw_bf  = (__bf16*)wsalloc(ws, nExpW * 2);
  float*  x       = (float*)wsalloc(ws, (size_t)S * Dm * 4);
  __bf16* xb      = (__bf16*)wsalloc(ws, (size_t)S * Dm * 2);
  __bf16* qkvb    = (__bf16*)wsalloc(ws, (size_t)S * 3 * Dm * 2);  // bf16 QKV (bias folded)
  float*  obuf    = (float*)wsalloc(ws, (size_t)S * Dm * 4);
  float*  oproj   = (float*)wsalloc(ws, (size_t)S * Dm * 4);
  float*  moe     = (float*)wsalloc(ws, (size_t)S * Dm * 4);
  int*    topi    = (int*)wsalloc(ws, (size_t)2 * S * 4);
  float*  topw    = (float*)wsalloc(ws, (size_t)2 * S * 4);
  int*    list    = (int*)wsalloc(ws, (size_t)Ee * S * 4);
  float*  listw   = (float*)wsalloc(ws, (size_t)Ee * S * 4);
  int*    count   = (int*)wsalloc(ws, (size_t)Ee * 4);

  // 1) weights -> bf16 (entire bf16 weight set ~175 MB fits in the 192 MB L2)
  f2bf_kernel<<<4096, 256, 0, stream>>>(emb, emb_bf, nEmb);
  f2bf_kernel<<<2048, 256, 0, stream>>>(qkv_w, qkvw_bf, nQkvW);
  f2bf_kernel<<<1024, 256, 0, stream>>>(out_w, outw_bf, nOutW);
  f2bf_kernel<<<4096, 256, 0, stream>>>(eg_w, egw_bf, nExpW);
  f2bf_kernel<<<4096, 256, 0, stream>>>(eu_w, euw_bf, nExpW);
  f2bf_kernel<<<4096, 256, 0, stream>>>(ed_w, edw_bf, nExpW);

  // 2) embedding
  embed_kernel<<<S, 256, 0, stream>>>(tokens, emb, x, xb);

  for (int l = 0; l < Ll; ++l) {
    // QKV projection -> bf16 with bias folded (feeds attention WMMA directly, no cvt in inner loop)
    gemm_kernel<false, true><<<dim3(3 * Dm / 64, S / 64), 256, 0, stream>>>(
        xb, qkvw_bf + (size_t)l * 3 * Dm * Dm, qkv_b + (size_t)l * 3 * Dm, qkvb, S, 3 * Dm, Dm);
    // flash attention, 1 wave / (16q-tile, head)
    attn_kernel<<<(S / 16) * Hh / 8, 256, 0, stream>>>(qkvb, obuf);
    // out projection (f32 A path)
    gemm_kernel<true, false><<<dim3(Dm / 64, S / 64), 256, 0, stream>>>(
        obuf, outw_bf + (size_t)l * Dm * Dm, out_b + (size_t)l * Dm, oproj, S, Dm, Dm);
    // residual + LN1
    add_ln_kernel<<<S, 256, 0, stream>>>(x, oproj, ln1_w + (size_t)l * Dm, ln1_b + (size_t)l * Dm, xb);
    // MoE: gate -> deterministic expert lists -> grouped WMMA FFN (top-2 only: 4x fewer FLOPs than dense)
    gate_topk_kernel<<<S / 8, 256, 0, stream>>>(x, gate_w + (size_t)l * Ee * Dm, topi, topw);
    build_lists_kernel<<<Ee, 256, 0, stream>>>(topi, topw, list, listw, count);
    hipMemsetAsync(moe, 0, (size_t)S * Dm * 4, stream);
    moe_ffn_kernel<<<dim3(S / 16, Ee), 256, 0, stream>>>(
        xb, egw_bf + (size_t)l * Ee * Ff * Dm, euw_bf + (size_t)l * Ee * Ff * Dm,
        edw_bf + (size_t)l * Ee * Dm * Ff, list, listw, count, moe);
    // residual + LN2
    add_ln_kernel<<<S, 256, 0, stream>>>(x, moe, ln2_w + (size_t)l * Dm, ln2_b + (size_t)l * Dm, xb);
  }

  // final RMSNorm + tied LM head with TDM-staged weight tiles (96 KB dynamic LDS per block)
  rms_kernel<<<S, 256, 0, stream>>>(x, rms_w, xb);
  lmhead_kernel<<<dim3(Vv / 64, S / 64), 256, 64 * Dm * sizeof(__bf16), stream>>>(
      xb, emb_bf, logits);
}
